// RecurrentGCN_50972671868972
// MI455X (gfx1250) — matrix operand order repaired and checked by reading, not compiled, not run on hardware
//
#include <hip/hip_runtime.h>
#include <hip/hip_bf16.h>

typedef __bf16 v16bf __attribute__((ext_vector_type(16)));
typedef float  v8f   __attribute__((ext_vector_type(8)));

#define N_NODES 100000
#define F_IN    256
#define HID     32
#define IN_CAT  288   // 256 + 32

// d_ws layout (unsigned short units):
//   [0      .. 16383]  : W fragments hi   (8 s * 4 t * 32 lanes * 16 halves)
//   [16384  .. 32767]  : W fragments lo
//   [32768  .. 33791]  : fc1 fragments hi (2 t * 32 lanes * 16 halves)
//   [33792  .. 34815]  : fc1 fragments lo
// total 69632 bytes

// Native hw converts (v_cvt_*bf16*): hi = rne(f), lo = rne(f - float(hi))
__device__ __forceinline__ void split_bf16(float f, __bf16& hi, __bf16& lo) {
  __bf16 h = (__bf16)f;
  hi = h;
  lo = (__bf16)(f - (float)h);   // (float)__bf16 is exact (shift)
}

// B-matrix (KxN, 16-bit) lane layout: lane holds column N=lane%16,
// halves i=0..7 map to K = i + 8*(lane/16), i=8..15 to K = 16 + (i-8) + 8*(lane/16).
__device__ __forceinline__ int klocalB(int lane, int i) {
  return i + ((lane < 16) ? 0 : 8) + ((i < 8) ? 0 : 8);
}

// ---------------- prep: build bf16 hi/lo B-fragments in d_ws ----------------
__global__ __launch_bounds__(256) void prep_kernel(
    const float* __restrict__ wz, const float* __restrict__ wh,
    const float* __restrict__ fc1w,
    __bf16* __restrict__ whi, __bf16* __restrict__ wlo,
    __bf16* __restrict__ f1hi, __bf16* __restrict__ f1lo) {
  int e = blockIdx.x * blockDim.x + threadIdx.x;
  if (e < 8 * 4 * 32 * 16) {
    // combined weight Wc[256][64]: cols 0..31 = Wz, 32..63 = Wh (K<256 rows only)
    int i = e & 15, lane = (e >> 4) & 31, t = (e >> 9) & 3, s = e >> 11;
    int k = 32 * s + klocalB(lane, i);
    int col = ((t & 1) << 4) + (lane & 15);
    const float* w = (t < 2) ? wz : wh;           // shape (2, 288, 32)
    float v = w[k * HID + col] + w[IN_CAT * HID + k * HID + col];
    __bf16 hi, lo; split_bf16(v, hi, lo);
    whi[e] = hi; wlo[e] = lo;
  } else if (e < 8 * 4 * 32 * 16 + 2 * 32 * 16) {
    int e2 = e - 8 * 4 * 32 * 16;
    int i = e2 & 15, lane = (e2 >> 4) & 31, t = e2 >> 9;
    int k = klocalB(lane, i);                     // K = 0..31
    int col = (t << 4) + (lane & 15);
    float v = fc1w[k * HID + col];                // fc1_w shape [32 in, 32 out]
    __bf16 hi, lo; split_bf16(v, hi, lo);
    f1hi[e2] = hi; f1lo[e2] = lo;
  }
}

// wave-internal LDS ordering fence (producer/consumer lanes within one wave)
__device__ __forceinline__ void wave_lds_fence() {
  __builtin_amdgcn_wave_barrier();
  asm volatile("s_wait_dscnt 0" ::: "memory");
  __builtin_amdgcn_wave_barrier();
}

__device__ __forceinline__ v8f wmma_bf16(v16bf a, v16bf b, v8f c) {
  return __builtin_amdgcn_wmma_f32_16x16x32_bf16(
      /*neg_a=*/false, a, /*neg_b=*/false, b,
      /*c_mod=*/(short)0, c, /*reuse_a=*/false, /*reuse_b=*/false);
}

// ---------------- main fused kernel ----------------
__global__ __launch_bounds__(256) void dcrnn_kernel(
    const float* __restrict__ x,
    const float* __restrict__ bz, const float* __restrict__ bh,
    const float* __restrict__ fc1b,
    const float* __restrict__ fc2w, const float* __restrict__ fc2b,
    const __bf16* __restrict__ wsfrag,
    float* __restrict__ out) {
  __shared__ v16bf wfrag[2048];        // 64 KB: [0..1023]=hi frags, [1024..2047]=lo
  __shared__ float hscratch[8][512];   // 16 KB: per-wave 16x32 transpose patch

  const int tid = threadIdx.x;
  {  // block-cooperative stage of all B fragments (hi then lo, contiguous in ws)
    const uint4* __restrict__ src = (const uint4*)wsfrag;
    uint4* dst = (uint4*)wfrag;
    #pragma unroll
    for (int i = 0; i < 16; ++i) dst[tid + 256 * i] = src[tid + 256 * i];
  }
  __syncthreads();

  const int wave = tid >> 5;
  const int lane = tid & 31;
  const int base = (blockIdx.x * 8 + wave) * 16;
  if (base >= N_NODES) return;        // N_NODES % 16 == 0 -> whole tiles only

  const int c = lane & 15;            // column / row-in-tile id
  const int hiHalf = lane >> 4;       // 0: lanes 0-15, 1: lanes 16-31
  const float* __restrict__ xrow = x + (size_t)(base + c) * F_IN;

  // ---- big GEMM: x_tile[16,256] @ Wc[256,64], bf16 hi/lo split, f32 acc ----
  v8f acc[4] = {{}, {}, {}, {}};
  #pragma unroll
  for (int s = 0; s < 8; ++s) {
    // A-fragment (16-bit A 16x32 layout): lane row = c; halves 0-7 take
    // K = k0..k0+7, halves 8-15 take K = k0+16..k0+23, k0 = 32s + 8*hiHalf
    const int k0 = 32 * s + (hiHalf ? 8 : 0);
    const float4 c0 = *(const float4*)(xrow + k0);
    const float4 c1 = *(const float4*)(xrow + k0 + 4);
    const float4 c2 = *(const float4*)(xrow + k0 + 16);
    const float4 c3 = *(const float4*)(xrow + k0 + 20);
    const float f[16] = {c0.x, c0.y, c0.z, c0.w, c1.x, c1.y, c1.z, c1.w,
                         c2.x, c2.y, c2.z, c2.w, c3.x, c3.y, c3.z, c3.w};
    v16bf ahi = {}, alo = {};
    #pragma unroll
    for (int i = 0; i < 16; ++i) {
      __bf16 h, l; split_bf16(f[i], h, l);
      ahi[i] = h; alo[i] = l;
    }
    #pragma unroll
    for (int t = 0; t < 4; ++t) {     // t0/t1: z cols 0-15/16-31; t2/t3: h-pre
      const v16bf bhi = wfrag[(s * 4 + t) * 32 + lane];
      const v16bf blo = wfrag[1024 + (s * 4 + t) * 32 + lane];
      acc[t] = wmma_bf16(ahi, bhi, acc[t]);
      acc[t] = wmma_bf16(alo, bhi, acc[t]);   // lo*hi correction
      acc[t] = wmma_bf16(ahi, blo, acc[t]);   // hi*lo correction
    }
  }

  // ---- GRU combine in C layout, write h to LDS for transpose ----
  // C layout: acc[t][j] = row (j + 8*hiHalf), col (16*(t&1) + c)
  const float bz0 = bz[c], bz1 = bz[c + 16];
  const float bh0 = bh[c], bh1 = bh[c + 16];
  const int rowoff = hiHalf * 8;
  float* hw = hscratch[wave];
  #pragma unroll
  for (int j = 0; j < 8; ++j) {
    float z0 = 1.f / (1.f + __expf(-(acc[0][j] + bz0)));
    float z1 = 1.f / (1.f + __expf(-(acc[1][j] + bz1)));
    float t0 = tanhf(acc[2][j] + bh0);
    float t1 = tanhf(acc[3][j] + bh1);
    hw[(rowoff + j) * 32 + c]      = (1.f - z0) * t0;   // h = (1-z)*h_tilde
    hw[(rowoff + j) * 32 + c + 16] = (1.f - z1) * t1;
  }
  wave_lds_fence();

  // ---- h[16,32] @ fc1[32,32] via WMMA (A-frag of h read back transposed) ----
  const float* hrow = hw + c * 32;
  const int p0 = hiHalf ? 8 : 0;
  const float4 q0 = *(const float4*)(hrow + p0);
  const float4 q1 = *(const float4*)(hrow + p0 + 4);
  const float4 q2 = *(const float4*)(hrow + p0 + 16);
  const float4 q3 = *(const float4*)(hrow + p0 + 20);
  const float g[16] = {q0.x, q0.y, q0.z, q0.w, q1.x, q1.y, q1.z, q1.w,
                       q2.x, q2.y, q2.z, q2.w, q3.x, q3.y, q3.z, q3.w};
  v16bf hhi = {}, hlo = {};
  #pragma unroll
  for (int i = 0; i < 16; ++i) {
    __bf16 h, l; split_bf16(g[i], h, l);
    hhi[i] = h; hlo[i] = l;
  }
  const v16bf* __restrict__ f1 = (const v16bf*)(wsfrag + 32768); // fc1 frags
  v8f hacc[2] = {{}, {}};
  #pragma unroll
  for (int t = 0; t < 2; ++t) {
    const v16bf fhi = f1[t * 32 + lane];
    const v16bf flo = f1[64 + t * 32 + lane];
    hacc[t] = wmma_bf16(hhi, fhi, hacc[t]);
    hacc[t] = wmma_bf16(hlo, fhi, hacc[t]);
    hacc[t] = wmma_bf16(hhi, flo, hacc[t]);
  }

  // ---- bias + relu, stash hidden back into the same LDS patch ----
  const float b10 = fc1b[c], b11 = fc1b[c + 16];
  wave_lds_fence();   // all lanes done reading h before overwrite
  #pragma unroll
  for (int j = 0; j < 8; ++j) {
    float u0 = hacc[0][j] + b10; u0 = u0 > 0.f ? u0 : 0.f;
    float u1 = hacc[1][j] + b11; u1 = u1 > 0.f ? u1 : 0.f;
    hw[(rowoff + j) * 32 + c]      = u0;
    hw[(rowoff + j) * 32 + c + 16] = u1;
  }
  wave_lds_fence();

  // ---- fc2 (32 -> 1) + sigmoid: one output row per lane 0..15 ----
  if (lane < 16) {
    float d = fc2b[0];
    const float* hr = hw + lane * 32;
    #pragma unroll
    for (int k = 0; k < 32; ++k) d += hr[k] * fc2w[k];
    out[base + lane] = 1.f / (1.f + __expf(-d));
  }
}

extern "C" void kernel_launch(void* const* d_in, const int* in_sizes, int n_in,
                              void* d_out, int out_size, void* d_ws, size_t ws_size,
                              hipStream_t stream) {
  const float* x    = (const float*)d_in[0];
  // d_in[1] edge_index: unused (K=1 DConv does no propagation)
  const float* wz   = (const float*)d_in[2];
  const float* bz   = (const float*)d_in[3];
  // d_in[4], d_in[5] (w_r, b_r): unused since r * h0 == 0
  const float* wh   = (const float*)d_in[6];
  const float* bh   = (const float*)d_in[7];
  const float* fc1w = (const float*)d_in[8];
  const float* fc1b = (const float*)d_in[9];
  const float* fc2w = (const float*)d_in[10];
  const float* fc2b = (const float*)d_in[11];
  __bf16* ws = (__bf16*)d_ws;

  // 17408 fragment elements total -> 68 blocks of 256
  prep_kernel<<<68, 256, 0, stream>>>(wz, wh, fc1w,
                                      ws, ws + 16384,
                                      ws + 32768, ws + 32768 + 1024);

  const int tiles = N_NODES / 16;            // 6250
  const int blocks = (tiles + 7) / 8;        // 782, 8 waves (tiles) per block
  dcrnn_kernel<<<blocks, 256, 0, stream>>>(x, bz, bh, fc1b, fc2w, fc2b, ws,
                                           (float*)d_out);
}
